// StripedLSTM_16621523435910
// MI455X (gfx1250) — matrix operand order repaired and checked by reading, compiled, not run on hardware
//
#include <hip/hip_runtime.h>

// ---------------------------------------------------------------------------
// Striped LSTM for MI455X (gfx1250, wave32, WMMA bf16 16x16x32)
//   L=2 layers, T=512, B=32, IN=H=1024, S=8 stripes of SS=128
// Pipeline:
//   (0) convert_kernel  : one-time fp32 -> bf16 for x and W_ih (halves GEMM
//       global traffic; removes per-tile VALU converts from the hot loop)
//   (1) gemm_xw_kernel  : xW = A_bf16 @ W_bf16^T + bias. 128x128 block,
//       BK=64, double-buffered LDS staged with global_load_async_to_lds_b128
//       (ASYNCcnt-pipelined), 16 v_wmma per wave per barrier.
//   (2) lstm_rec_kernel : persistent per-(stripe, batch-half) workgroups,
//       W_hh resident in LDS (bf16, 128KB), h in LDS (bf16), c in regs.
//       Writes bf16 seq for the next layer's GEMM (fp32 only for final layer).
// ---------------------------------------------------------------------------

#define L_    2
#define S_    8
#define SSZ   128
#define H_    1024
#define IN_   1024
#define T_    512
#define B_    32
#define G4    512              // 4*SSZ gates per stripe
#define NGATE 4096             // S_*G4
#define MROWS (T_*B_)          // 16384

typedef __attribute__((ext_vector_type(16))) __bf16 v16bf;
typedef __attribute__((ext_vector_type(8)))  __bf16 v8bf;
typedef __attribute__((ext_vector_type(8)))  float  v8f;

__device__ __forceinline__ __bf16 to_bf16(float f) {
  union { float f; unsigned u; } x; x.f = f;
  unsigned r = (x.u + 0x7FFFu + ((x.u >> 16) & 1u)) >> 16;   // RNE
  union { unsigned short s; __bf16 b; } y; y.s = (unsigned short)r;
  return y.b;
}

// LDS-relative byte offset of a pointer into a __shared__ array.
__device__ __forceinline__ unsigned lds_off(const void* p) {
  return (unsigned)(size_t)(const __attribute__((address_space(3))) void*)p;
}

// One 64-byte segment (32 bf16) per lane: 4 async b128 transfers.
// INST_OFFSET is added to both LDS and global addresses (ISA 15.18.3).
__device__ __forceinline__ void async_copy64(unsigned ldsa, unsigned va,
                                             unsigned long long base) {
  asm volatile("global_load_async_to_lds_b128 %0, %1, %2 offset:0"
               :: "v"(ldsa), "v"(va), "s"(base) : "memory");
  asm volatile("global_load_async_to_lds_b128 %0, %1, %2 offset:16"
               :: "v"(ldsa), "v"(va), "s"(base) : "memory");
  asm volatile("global_load_async_to_lds_b128 %0, %1, %2 offset:32"
               :: "v"(ldsa), "v"(va), "s"(base) : "memory");
  asm volatile("global_load_async_to_lds_b128 %0, %1, %2 offset:48"
               :: "v"(ldsa), "v"(va), "s"(base) : "memory");
}

// A operand (16x32 bf16): lane holds row (lane&15);
//   j=0..15 -> k = half*16 + j
__device__ __forceinline__ v16bf frag_a(const __bf16* tile, int ldk) {
  int lane = threadIdx.x & 31;
  int row  = lane & 15;
  int half = lane >> 4;
  const v8bf* p = (const v8bf*)(tile + row * ldk + half * 16);
  v8bf lo = p[0];
  v8bf hi = p[1];
  return __builtin_shufflevector(lo, hi, 0,1,2,3,4,5,6,7,8,9,10,11,12,13,14,15);
}

// B operand (32x16 bf16), B[k][n] = W[n][k]: lane holds column n = lane&15,
// k = half*16 .. +15 contiguous.
__device__ __forceinline__ v16bf frag_b(const __bf16* tile, int ldk) {
  int lane = threadIdx.x & 31;
  int col  = lane & 15;
  int half = lane >> 4;
  const v8bf* p = (const v8bf*)(tile + col * ldk + half * 16);
  v8bf lo = p[0];
  v8bf hi = p[1];
  return __builtin_shufflevector(lo, hi, 0,1,2,3,4,5,6,7,8,9,10,11,12,13,14,15);
}

__device__ __forceinline__ v8f wmma_bf16(v16bf a, v16bf b, v8f c) {
  return __builtin_amdgcn_wmma_f32_16x16x32_bf16(false, a, false, b,
                                                 (short)0, c, false, false);
}

// ---------------------------------------------------------------------------
// Kernel 0: elementwise fp32 -> bf16
// ---------------------------------------------------------------------------
__global__ __launch_bounds__(256) void convert_kernel(
    const float* __restrict__ src, __bf16* __restrict__ dst, int n4) {
  int i = (blockIdx.x * 256 + threadIdx.x);
  if (i < n4) {
    float4 v = ((const float4*)src)[i];
    __bf16* d = dst + i * 4;
    d[0] = to_bf16(v.x); d[1] = to_bf16(v.y);
    d[2] = to_bf16(v.z); d[3] = to_bf16(v.w);
  }
}

// ---------------------------------------------------------------------------
// Kernel 1: C[M=16384, N=4096] = A[M,K=1024]_bf16 * W[N,K]_bf16^T + bias[N]
// Block tile 128x128, BK=64, double-buffered async LDS staging.
// 256 threads = 8 waves, wave tile 64x32 (4 Mtiles x 2 Ntiles).
// ---------------------------------------------------------------------------
#define BM 128
#define BN 128
#define BK 64
#define NCH (IN_ / BK)   // 16 K-chunks

__global__ __launch_bounds__(256) void gemm_xw_kernel(
    const __bf16* __restrict__ A,   // M x K  row-major bf16
    const __bf16* __restrict__ W,   // N x K  row-major bf16
    const float* __restrict__ bih,  // N
    const float* __restrict__ bhh,  // N
    float* __restrict__ C)          // M x N  row-major fp32
{
  __shared__ __bf16 As[2][BM * BK];   // 2 x 16 KB
  __shared__ __bf16 Ws[2][BN * BK];   // 2 x 16 KB

  const int tid = threadIdx.x;
  const int m0  = blockIdx.y * BM;
  const int n0  = blockIdx.x * BN;
  const int w   = tid >> 5;
  const int wm  = w >> 2;           // 0..1 : M offset wm*64
  const int wn  = w & 3;            // 0..3 : N offset wn*32

  v8f acc[4][2];
#pragma unroll
  for (int i = 0; i < 4; ++i)
#pragma unroll
    for (int j = 0; j < 2; ++j)
      acc[i][j] = (v8f){0.f,0.f,0.f,0.f,0.f,0.f,0.f,0.f};

  // staging: thread owns row tid>>1 (0..127), 32-elem segment (tid&1)*32
  const int srow = tid >> 1;
  const int sseg = (tid & 1) * 32;
  const unsigned ldsA0 = lds_off(&As[0][srow * BK + sseg]);
  const unsigned ldsW0 = lds_off(&Ws[0][srow * BK + sseg]);
  const unsigned bufStride = (unsigned)(BM * BK * 2);   // 16 KB
  const unsigned gA0 = (unsigned)(((m0 + srow) * IN_ + sseg) * 2);
  const unsigned gW0 = (unsigned)(((n0 + srow) * IN_ + sseg) * 2);
  const unsigned long long aBase = (unsigned long long)(size_t)A;
  const unsigned long long wBase = (unsigned long long)(size_t)W;

  // prologue: chunk 0 -> buffer 0
  async_copy64(ldsA0, gA0, aBase);
  async_copy64(ldsW0, gW0, wBase);

  for (int ch = 0; ch < NCH; ++ch) {
    if (ch + 1 < NCH) {
      const unsigned koff = (unsigned)((ch + 1) * BK * 2);
      const unsigned boff = ((ch + 1) & 1) ? bufStride : 0u;
      async_copy64(ldsA0 + boff, gA0 + koff, aBase);
      async_copy64(ldsW0 + boff, gW0 + koff, wBase);
      asm volatile("s_wait_asynccnt 0x8" ::: "memory");  // chunk ch complete
    } else {
      asm volatile("s_wait_asynccnt 0x0" ::: "memory");
    }
    __syncthreads();

    const __bf16* Ab = As[ch & 1];
    const __bf16* Wb = Ws[ch & 1];
#pragma unroll
    for (int kk = 0; kk < 2; ++kk) {
      v16bf bfr[2];
#pragma unroll
      for (int tn = 0; tn < 2; ++tn)
        bfr[tn] = frag_b(Wb + (wn * 32 + tn * 16) * BK + kk * 32, BK);
#pragma unroll
      for (int tm = 0; tm < 4; ++tm) {
        v16bf afr = frag_a(Ab + (wm * 64 + tm * 16) * BK + kk * 32, BK);
#pragma unroll
        for (int tn = 0; tn < 2; ++tn)
          acc[tm][tn] = wmma_bf16(afr, bfr[tn], acc[tm][tn]);
      }
    }
    __syncthreads();   // done reading this buffer before it is refilled
  }

  // epilogue: + bias, store fp32
  const int lane = tid & 31;
  const int col  = lane & 15;
  const int half = lane >> 4;
#pragma unroll
  for (int tm = 0; tm < 4; ++tm) {
#pragma unroll
    for (int tn = 0; tn < 2; ++tn) {
      int n = n0 + wn * 32 + tn * 16 + col;
      float bias = bih[n] + bhh[n];
#pragma unroll
      for (int v = 0; v < 8; ++v) {
        int m = m0 + wm * 64 + tm * 16 + half * 8 + v;
        C[(size_t)m * NGATE + n] = acc[tm][tn][v] + bias;
      }
    }
  }
}

// ---------------------------------------------------------------------------
// Kernel 2: persistent recurrence. Grid = 16: blockIdx = stripe*2 + batchHalf.
// 256 threads = 8 waves; wave w owns gate columns [w*64, w*64+64).
// LDS: W_hh stripe bf16 (128KB) + h bf16 (4KB) + gates fp32 (32KB) = 164KB.
// ---------------------------------------------------------------------------
#define REC_LDS_BYTES (G4*SSZ*2 + 16*SSZ*2 + 16*G4*4)

__global__ __launch_bounds__(256) void lstm_rec_kernel(
    const float* __restrict__ xw,      // (T*B) x 4096  (xW + bias)
    const float* __restrict__ Whh_l,   // S x 512 x 128 (this layer)
    __bf16* __restrict__ seq_bf,       // (T*B) x 1024 bf16, or nullptr
    float* __restrict__ seq_f32,       // (T*B) x 1024 fp32, or nullptr
    float* __restrict__ hn,            // B x 1024
    float* __restrict__ cn)            // B x 1024
{
  extern __shared__ char smem[];
  __bf16* Wl = (__bf16*)smem;                                // 512 x 128
  __bf16* hl = (__bf16*)(smem + G4 * SSZ * 2);               // 16 x 128
  float*  gl = (float*)(smem + G4 * SSZ * 2 + 16 * SSZ * 2); // 16 x 512

  const int tid = threadIdx.x;
  const int s   = blockIdx.x >> 1;
  const int bh  = blockIdx.x & 1;          // batch half: rows bh*16 .. +15

  // stage W_hh[s] into LDS as bf16 (gate-row major, K contiguous)
  const float* Wsrc = Whh_l + (size_t)s * G4 * SSZ;
  for (int i = tid * 4; i < G4 * SSZ; i += 256 * 4) {
    float4 v = *(const float4*)(Wsrc + i);
    Wl[i + 0] = to_bf16(v.x); Wl[i + 1] = to_bf16(v.y);
    Wl[i + 2] = to_bf16(v.z); Wl[i + 3] = to_bf16(v.w);
  }
  for (int i = tid; i < 16 * SSZ; i += 256) hl[i] = to_bf16(0.0f);

  const int w      = tid >> 5;
  const int n_base = w * 64;
  const int lane   = tid & 31;
  const int col    = lane & 15;
  const int half   = lane >> 4;

  // activation cells: 8 consecutive j of one batch row per thread
  const int cell0 = tid * 8;
  const int bcell = cell0 >> 7;       // 0..15 local batch row
  const int j0    = cell0 & 127;

  float c_reg[8], h_reg[8];
#pragma unroll
  for (int i = 0; i < 8; ++i) { c_reg[i] = 0.f; h_reg[i] = 0.f; }

  for (int t = 0; t < T_; ++t) {
    const size_t row = (size_t)(t * B_ + bh * 16 + bcell);
    const float* xwrow = xw + row * NGATE + s * G4 + j0;
    // warm caches for the gate segments read after the WMMAs
    __builtin_prefetch(xwrow, 0, 0);
    __builtin_prefetch(xwrow + 128, 0, 0);
    __builtin_prefetch(xwrow + 256, 0, 0);
    __builtin_prefetch(xwrow + 384, 0, 0);

    __syncthreads();   // h writes from previous step complete

    v8f acc[4];
#pragma unroll
    for (int tn = 0; tn < 4; ++tn)
      acc[tn] = (v8f){0.f,0.f,0.f,0.f,0.f,0.f,0.f,0.f};

#pragma unroll
    for (int kk = 0; kk < 4; ++kk) {
      v16bf a = frag_a(hl + kk * 32, SSZ);
#pragma unroll
      for (int tn = 0; tn < 4; ++tn) {
        v16bf b = frag_b(Wl + (size_t)(n_base + tn * 16) * SSZ + kk * 32, SSZ);
        acc[tn] = wmma_bf16(a, b, acc[tn]);
      }
    }
#pragma unroll
    for (int tn = 0; tn < 4; ++tn)
#pragma unroll
      for (int v = 0; v < 8; ++v)
        gl[(half * 8 + v) * G4 + n_base + tn * 16 + col] = acc[tn][v];

    __syncthreads();   // gates visible

    const float* grow = gl + bcell * G4 + j0;
    float*  of = seq_f32 ? seq_f32 + row * H_ + s * SSZ + j0 : nullptr;
    __bf16* ob = seq_bf  ? seq_bf  + row * H_ + s * SSZ + j0 : nullptr;
#pragma unroll
    for (int i = 0; i < 8; ++i) {
      float gi = grow[i]       + xwrow[i];
      float gf = grow[128 + i] + xwrow[128 + i];
      float gg = grow[256 + i] + xwrow[256 + i];
      float go = grow[384 + i] + xwrow[384 + i];
      float si = 1.f / (1.f + __expf(-gi));
      float sf = 1.f / (1.f + __expf(-gf));
      float so = 1.f / (1.f + __expf(-go));
      float c  = sf * c_reg[i] + si * tanhf(gg);
      float h  = so * tanhf(c);
      c_reg[i] = c;
      h_reg[i] = h;
      __bf16 hb = to_bf16(h);
      hl[bcell * SSZ + j0 + i] = hb;
      if (of) of[i] = h;
      if (ob) ob[i] = hb;
    }
  }

  float* hrow = hn + (size_t)(bh * 16 + bcell) * H_ + s * SSZ + j0;
  float* crow = cn + (size_t)(bh * 16 + bcell) * H_ + s * SSZ + j0;
#pragma unroll
  for (int i = 0; i < 8; ++i) { hrow[i] = h_reg[i]; crow[i] = c_reg[i]; }
}

// ---------------------------------------------------------------------------
extern "C" void kernel_launch(void* const* d_in, const int* in_sizes, int n_in,
                              void* d_out, int out_size, void* d_ws, size_t ws_size,
                              hipStream_t stream) {
  const float* x    = (const float*)d_in[0];   // T*B*IN
  const float* W_ih = (const float*)d_in[1];   // L*S*512*IN
  const float* W_hh = (const float*)d_in[2];   // L*S*512*128
  const float* b_ih = (const float*)d_in[3];   // L*4096
  const float* b_hh = (const float*)d_in[4];   // L*4096

  float* out     = (float*)d_out;
  float* seq_out = out;                           // T*B*H
  float* hn      = out + (size_t)MROWS * H_;      // L*B*H
  float* cn      = hn + (size_t)L_ * B_ * H_;     // L*B*H

  // workspace layout
  float*  xwbuf = (float*)d_ws;                              // 16384x4096 f32
  __bf16* xbf   = (__bf16*)(xwbuf + (size_t)MROWS * NGATE);  // 16384x1024 bf16
  __bf16* seqbf = xbf + (size_t)MROWS * IN_;                 // 16384x1024 bf16
  __bf16* wbf   = seqbf + (size_t)MROWS * H_;                // 2x4096x1024 bf16

  (void)in_sizes; (void)n_in; (void)out_size; (void)ws_size;

  hipFuncSetAttribute(reinterpret_cast<const void*>(lstm_rec_kernel),
                      hipFuncAttributeMaxDynamicSharedMemorySize, REC_LDS_BYTES);

  // one-time fp32 -> bf16 of x and both layers' W_ih
  {
    int n4 = (MROWS * IN_) / 4;
    convert_kernel<<<(n4 + 255) / 256, 256, 0, stream>>>(x, xbf, n4);
    int w4 = (L_ * NGATE * IN_) / 4;
    convert_kernel<<<(w4 + 255) / 256, 256, 0, stream>>>(W_ih, wbf, w4);
  }

  dim3 ggrid(NGATE / BN, MROWS / BM);   // 32 x 128

  // ---- layer 0 ----
  gemm_xw_kernel<<<ggrid, 256, 0, stream>>>(xbf, wbf, b_ih, b_hh, xwbuf);
  lstm_rec_kernel<<<16, 256, REC_LDS_BYTES, stream>>>(
      xwbuf, W_hh, seqbf, nullptr, hn, cn);

  // ---- layer 1 ----
  const __bf16* wbf1 = wbf + (size_t)NGATE * IN_;
  const float*  Whh1 = W_hh + (size_t)S_ * G4 * SSZ;
  gemm_xw_kernel<<<ggrid, 256, 0, stream>>>(seqbf, wbf1, b_ih + NGATE,
                                            b_hh + NGATE, xwbuf);
  lstm_rec_kernel<<<16, 256, REC_LDS_BYTES, stream>>>(
      xwbuf, Whh1, nullptr, seq_out, hn + (size_t)B_ * H_, cn + (size_t)B_ * H_);
}
